// AttentionOp_3350074490884
// MI455X (gfx1250) — matrix operand order repaired and checked by reading, not compiled, not run on hardware
//
#include <hip/hip_runtime.h>
#include <stdint.h>

typedef __attribute__((ext_vector_type(16))) _Float16 v16h;
typedef __attribute__((ext_vector_type(8)))  _Float16 v8h;
typedef __attribute__((ext_vector_type(4)))  _Float16 v4h;
typedef __attribute__((ext_vector_type(8)))  float    v8f;
typedef __attribute__((ext_vector_type(4)))  float    v4f;

union V16 { v16h v; v8h h[2]; };

#define B_   2
#define S_   2048
#define HQ   16
#define HKV  8
#define D_   128
#define C_   512
#define NCH  4

#define KPAD 136   // 128 + 8 f16 row pad
#define VPAD 40    // 32 + 8 f16 row pad

#define KV16_ELEMS ((size_t)B_ * NCH * HKV * C_ * D_)   // 4,194,304 halves = 8 MB

__device__ __forceinline__ float rowmax16(float x) {
  x = fmaxf(x, __shfl_xor(x, 1, 32));
  x = fmaxf(x, __shfl_xor(x, 2, 32));
  x = fmaxf(x, __shfl_xor(x, 4, 32));
  x = fmaxf(x, __shfl_xor(x, 8, 32));
  return x;
}
__device__ __forceinline__ float rowsum16(float x) {
  x += __shfl_xor(x, 1, 32);
  x += __shfl_xor(x, 2, 32);
  x += __shfl_xor(x, 4, 32);
  x += __shfl_xor(x, 8, 32);
  return x;
}

// async global->LDS copy of 16 bytes (CDNA5 GLOBAL_LOAD_ASYNC_TO_LDS_B128, ASYNCcnt)
__device__ __forceinline__ void async_cp16(unsigned lds_off, const void* gaddr) {
  asm volatile("global_load_async_to_lds_b128 %0, %1, off"
               :: "v"(lds_off), "v"(gaddr) : "memory");
}
__device__ __forceinline__ unsigned lds_off32(const void* p) {
  // flat LDS addresses are {aperture_hi32, lds_offset32}; low 32 bits are the DS address
  return (unsigned)(uintptr_t)p;
}

// ============================================================================
// Prep kernel: K (f32) -> f16 row-major [b][nc][kvh][key][d]
//              V (f32) -> f16 TRANSPOSED [b][nc][kvh][d][key] (via LDS tile,
//              so global stores stay coalesced)
// One block per (b, nc, kvh, 128-key tile): grid = 2*4*8*4 = 256
// ============================================================================
__global__ __launch_bounds__(256) void prep_kv_f16(
    const float* __restrict__ kg, const float* __restrict__ vg,
    _Float16* __restrict__ k16, _Float16* __restrict__ vt16) {
  __shared__ __align__(16) _Float16 vt[128 * KPAD];

  int bid = blockIdx.x;
  const int kt  = bid & 3; bid >>= 2;   // 128-key tile
  const int kvh = bid & 7; bid >>= 3;
  const int nc  = bid & 3;
  const int b   = bid >> 2;
  const int tid = threadIdx.x;

  const size_t head = (size_t)((b * NCH + nc) * HKV + kvh);
  const float* kin = kg + (((size_t)b * S_ + (size_t)nc * C_ + kt * 128) * HKV + kvh) * D_;
  const float* vin = vg + (((size_t)b * S_ + (size_t)nc * C_ + kt * 128) * HKV + kvh) * D_;
  _Float16* kout  = k16  + head * (C_ * D_) + (size_t)kt * 128 * D_;
  _Float16* vtout = vt16 + head * (D_ * C_) + kt * 128;   // column offset in key dim

  // 128 keys x 128 d: per thread 16 float4 slots
  #pragma unroll
  for (int i = 0; i < 16; ++i) {
    const int slot = tid + i * 256;        // 0..4095
    const int row  = slot >> 5;            // key row in tile
    const int dc   = (slot & 31) * 4;      // d
    v4f kf = *(const v4f*)(kin + (size_t)row * (HKV * D_) + dc);
    v4f vf = *(const v4f*)(vin + (size_t)row * (HKV * D_) + dc);
    v4h kh;
    #pragma unroll
    for (int c = 0; c < 4; ++c) kh[c] = (_Float16)kf[c];
    *(v4h*)&kout[(size_t)row * D_ + dc] = kh;
    #pragma unroll
    for (int c = 0; c < 4; ++c) vt[(dc + c) * KPAD + row] = (_Float16)vf[c];
  }
  __syncthreads();
  // write transposed tile coalesced: 128 d rows x 128 keys, per thread 16 x 8-half slots
  #pragma unroll
  for (int i = 0; i < 16; ++i) {
    const int slot = tid + i * 256;        // 0..4095
    const int d    = slot >> 4;
    const int ko   = (slot & 15) * 8;
    v8h x = *(const v8h*)&vt[d * KPAD + ko];
    *(v8h*)&vtout[(size_t)d * C_ + ko] = x;
  }
}

// ============================================================================
// Attention kernel (async double-buffered): flash attention inside each chunk.
// K/V already f16 (V transposed), tiles DMA'd to LDS with async-to-LDS copies.
// Key-tile loop manually unrolled x2 so the LDS double-buffer index is a
// compile-time constant -> static ds offsets -> pipelined ds_load/wmma.
// ============================================================================
__global__ __launch_bounds__(256) void chunk_attn_wmma_async(
    const float* __restrict__ qg, const _Float16* __restrict__ k16,
    const _Float16* __restrict__ vt16, float* __restrict__ out) {
  __shared__ __align__(16) _Float16 sK[2][32 * KPAD];
  __shared__ __align__(16) _Float16 sVt[2][D_ * VPAD];
  __shared__ __align__(16) _Float16 sP[8 * 16 * 32];

  const int tid  = threadIdx.x;
  const int lane = tid & 31;
  const int wave = tid >> 5;
  const int hl   = lane >> 4;
  const int l16  = lane & 15;

  int bid = blockIdx.x;
  const int qblock = bid & 3;  bid >>= 2;
  const int hq     = bid & 15; bid >>= 4;
  const int nc     = bid & 3;
  const int b      = bid >> 2;
  const int kvh    = hq >> 1;

  const int qlo = qblock * 128 + wave * 16;
  const float scale = 0.08838834764831845f;  // 1/sqrt(128)

  const size_t head = (size_t)((b * NCH + nc) * HKV + kvh);
  const _Float16* kbase  = k16  + head * (C_ * D_);
  const _Float16* vtbase = vt16 + head * (D_ * C_);

  // per-thread async copy slots: K tile 32x128 f16 = 512 x 16B ; Vt tile 128x32 f16 = 512 x 16B
  auto issue_tile = [&](int jt, int buf) {
    const int klo = jt * 32;
    #pragma unroll
    for (int i = 0; i < 2; ++i) {
      const int slot = tid + i * 256;      // 0..511
      const int row  = slot >> 4;          // key row
      const int c16  = (slot & 15) * 8;    // d (halves)
      async_cp16(lds_off32(&sK[buf][row * KPAD + c16]),
                 kbase + (size_t)(klo + row) * D_ + c16);
    }
    #pragma unroll
    for (int i = 0; i < 2; ++i) {
      const int slot = tid + i * 256;      // 0..511
      const int d    = slot >> 2;          // d row
      const int kk   = (slot & 3) * 8;     // key (halves)
      async_cp16(lds_off32(&sVt[buf][d * VPAD + kk]),
                 vtbase + (size_t)d * C_ + klo + kk);
    }
  };

  issue_tile(0, 0);  // start DMA of first tile before anything else

  // ---- load Q tile (16 x 128) once, into WMMA A-operand layout (overlaps with DMA)
  const int qrow = qlo + l16;
  const float* qptr = qg + (((size_t)b * S_ + (size_t)nc * C_ + qrow) * HQ + hq) * D_;
  V16 qa[4];
  #pragma unroll
  for (int c = 0; c < 4; ++c) {
    const int base = c * 32 + hl * 8;
    v4f x0 = *(const v4f*)(qptr + base);
    v4f x1 = *(const v4f*)(qptr + base + 4);
    v4f y0 = *(const v4f*)(qptr + base + 16);
    v4f y1 = *(const v4f*)(qptr + base + 20);
    #pragma unroll
    for (int i = 0; i < 4; ++i) {
      qa[c].h[0][i]     = (_Float16)x0[i];
      qa[c].h[0][4 + i] = (_Float16)x1[i];
      qa[c].h[1][i]     = (_Float16)y0[i];
      qa[c].h[1][4 + i] = (_Float16)y1[i];
    }
  }

  v8f o[8];
  #pragma unroll
  for (int n = 0; n < 8; ++n) o[n] = (v8f){0.f,0.f,0.f,0.f,0.f,0.f,0.f,0.f};
  float mrow[8], lrow[8];
  #pragma unroll
  for (int r = 0; r < 8; ++r) { mrow[r] = -3.0e38f; lrow[r] = 0.0f; }

  const int jcount = (qblock + 1) * 4;   // always a multiple of 4
  _Float16* sPw = &sP[wave * 16 * 32];

  // one pipeline step; `cur` is passed as a literal so LDS offsets are static
  auto step = [&](int jt, int cur) {
    const int klo = jt * 32;
    if (jt + 1 < jcount) {
      issue_tile(jt + 1, cur ^ 1);                       // prefetch next tile (DMA)
      asm volatile("s_wait_asynccnt 0x4" ::: "memory");  // oldest 4 (tile jt) done
    } else {
      asm volatile("s_wait_asynccnt 0x0" ::: "memory");
    }
    __syncthreads();  // all waves' copies for tile jt visible

    const _Float16* sKc = &sK[cur][0];    // static after const-fold
    const _Float16* sVc = &sVt[cur][0];

    const int active = (klo <= qlo + 15) ? 1 : 0;
    if (__builtin_amdgcn_readfirstlane(active)) {
      // ---- S = Q x K^T : 16x32 logits as two 16x16 n-tiles, 4 WMMA each over D=128
      v8f s[2];
      #pragma unroll
      for (int nt = 0; nt < 2; ++nt) {
        const int keyl = nt * 16 + l16;
        v8f acc = (v8f){0.f,0.f,0.f,0.f,0.f,0.f,0.f,0.f};
        #pragma unroll
        for (int c = 0; c < 4; ++c) {
          const int dbase = c * 32 + hl * 8;
          V16 bb;
          bb.h[0] = *(const v8h*)&sKc[keyl * KPAD + dbase];
          bb.h[1] = *(const v8h*)&sKc[keyl * KPAD + dbase + 16];
          acc = __builtin_amdgcn_wmma_f32_16x16x32_f16(
              false, qa[c].v, false, bb.v, (short)0, acc, false, false);
        }
        s[nt] = acc;
      }
      // ---- scale + chunk-causal mask in C-layout
      #pragma unroll
      for (int r = 0; r < 8; ++r) {
        const int qv = qlo + r + 8 * hl;
        s[0][r] = ((klo + l16)      <= qv) ? s[0][r] * scale : -2.38e38f;
        s[1][r] = ((klo + 16 + l16) <= qv) ? s[1][r] * scale : -2.38e38f;
      }
      // ---- online softmax
      float alpha[8], p0[8], p1[8];
      #pragma unroll
      for (int r = 0; r < 8; ++r) {
        float mt = rowmax16(fmaxf(s[0][r], s[1][r]));
        float mn = fmaxf(mrow[r], mt);
        alpha[r] = __expf(mrow[r] - mn);
        p0[r] = __expf(s[0][r] - mn);
        p1[r] = __expf(s[1][r] - mn);
        lrow[r] = lrow[r] * alpha[r] + rowsum16(p0[r] + p1[r]);
        mrow[r] = mn;
      }
      #pragma unroll
      for (int n = 0; n < 8; ++n)
        #pragma unroll
        for (int r = 0; r < 8; ++r) o[n][r] *= alpha[r];

      // ---- stage P (16x32 f16) through per-wave LDS: C-layout -> A-layout
      #pragma unroll
      for (int r = 0; r < 8; ++r) {
        const int pr = r + 8 * hl;
        sPw[pr * 32 + l16]      = (_Float16)p0[r];
        sPw[pr * 32 + 16 + l16] = (_Float16)p1[r];
      }
      asm volatile("s_wait_dscnt 0x0" ::: "memory");
      V16 pa;
      {
        const int base = hl * 8;
        pa.h[0] = *(const v8h*)&sPw[l16 * 32 + base];
        pa.h[1] = *(const v8h*)&sPw[l16 * 32 + base + 16];
      }
      // ---- O += P x V : 8 n-tiles over D
      #pragma unroll
      for (int n = 0; n < 8; ++n) {
        const int dcol = n * 16 + l16;
        const int kb2  = hl * 8;
        V16 vb;
        vb.h[0] = *(const v8h*)&sVc[dcol * VPAD + kb2];
        vb.h[1] = *(const v8h*)&sVc[dcol * VPAD + kb2 + 16];
        o[n] = __builtin_amdgcn_wmma_f32_16x16x32_f16(
            false, pa.v, false, vb.v, (short)0, o[n], false, false);
      }
    }
    __syncthreads();  // compute done before this buffer is overwritten
  };

  for (int jt = 0; jt < jcount; jt += 2) {
    step(jt, 0);      // buffer 0: static LDS offsets
    step(jt + 1, 1);  // buffer 1: static LDS offsets
  }

  // ---- epilogue: out = O / l
  #pragma unroll
  for (int r = 0; r < 8; ++r) {
    const int srow = nc * C_ + qlo + r + 8 * hl;
    const float inv = 1.0f / lrow[r];
    float* op = out + (((size_t)b * S_ + srow) * HQ + hq) * D_;
    #pragma unroll
    for (int n = 0; n < 8; ++n) op[n * 16 + l16] = o[n][r] * inv;
  }
}

// ============================================================================
// Fallback fused kernel (round-1 version, f32 inputs, VGPR-staged LDS tiles)
// used only if the workspace is too small for the f16 K/Vt copies.
// ============================================================================
__global__ __launch_bounds__(256) void chunk_attn_wmma_fused(
    const float* __restrict__ qg, const float* __restrict__ kg,
    const float* __restrict__ vg, float* __restrict__ out) {
  __shared__ __align__(16) _Float16 sK[32 * KPAD];
  __shared__ __align__(16) _Float16 sVt[D_ * VPAD];
  __shared__ __align__(16) _Float16 sP[8 * 16 * 32];

  const int tid  = threadIdx.x;
  const int lane = tid & 31;
  const int wave = tid >> 5;
  const int hl   = lane >> 4;
  const int l16  = lane & 15;

  int bid = blockIdx.x;
  const int qblock = bid & 3;  bid >>= 2;
  const int hq     = bid & 15; bid >>= 4;
  const int nc     = bid & 3;
  const int b      = bid >> 2;
  const int kvh    = hq >> 1;

  const int qlo = qblock * 128 + wave * 16;
  const float scale = 0.08838834764831845f;

  const int qrow = qlo + l16;
  const float* qptr = qg + (((size_t)b * S_ + (size_t)nc * C_ + qrow) * HQ + hq) * D_;
  V16 qa[4];
  #pragma unroll
  for (int c = 0; c < 4; ++c) {
    const int base = c * 32 + hl * 8;
    v4f x0 = *(const v4f*)(qptr + base);
    v4f x1 = *(const v4f*)(qptr + base + 4);
    v4f y0 = *(const v4f*)(qptr + base + 16);
    v4f y1 = *(const v4f*)(qptr + base + 20);
    #pragma unroll
    for (int i = 0; i < 4; ++i) {
      qa[c].h[0][i]     = (_Float16)x0[i];
      qa[c].h[0][4 + i] = (_Float16)x1[i];
      qa[c].h[1][i]     = (_Float16)y0[i];
      qa[c].h[1][4 + i] = (_Float16)y1[i];
    }
  }

  v8f o[8];
  #pragma unroll
  for (int n = 0; n < 8; ++n) o[n] = (v8f){0.f,0.f,0.f,0.f,0.f,0.f,0.f,0.f};
  float mrow[8], lrow[8];
  #pragma unroll
  for (int r = 0; r < 8; ++r) { mrow[r] = -3.0e38f; lrow[r] = 0.0f; }

  const float* kbase = kg + (((size_t)b * S_ + (size_t)nc * C_) * HKV + kvh) * D_;
  const float* vbase = vg + (((size_t)b * S_ + (size_t)nc * C_) * HKV + kvh) * D_;

  const int jcount = (qblock + 1) * 4;
  _Float16* sPw = &sP[wave * 16 * 32];

  for (int jt = 0; jt < jcount; ++jt) {
    const int klo = jt * 32;
    __syncthreads();
    #pragma unroll
    for (int i = 0; i < 4; ++i) {
      const int fl = tid + i * 256;
      const int kr = fl >> 5;
      const int d  = (fl & 31) * 4;
      v4f kf = *(const v4f*)(kbase + (size_t)(klo + kr) * (HKV * D_) + d);
      v4f vf = *(const v4f*)(vbase + (size_t)(klo + kr) * (HKV * D_) + d);
      v4h kh;
      #pragma unroll
      for (int c = 0; c < 4; ++c) kh[c] = (_Float16)kf[c];
      *(v4h*)&sK[kr * KPAD + d] = kh;
      #pragma unroll
      for (int c = 0; c < 4; ++c) sVt[(d + c) * VPAD + kr] = (_Float16)vf[c];
    }
    __syncthreads();

    const int active = (klo <= qlo + 15) ? 1 : 0;
    if (__builtin_amdgcn_readfirstlane(active)) {
      v8f s[2];
      #pragma unroll
      for (int nt = 0; nt < 2; ++nt) {
        const int keyl = nt * 16 + l16;
        v8f acc = (v8f){0.f,0.f,0.f,0.f,0.f,0.f,0.f,0.f};
        #pragma unroll
        for (int c = 0; c < 4; ++c) {
          const int dbase = c * 32 + hl * 8;
          V16 bb;
          bb.h[0] = *(const v8h*)&sK[keyl * KPAD + dbase];
          bb.h[1] = *(const v8h*)&sK[keyl * KPAD + dbase + 16];
          acc = __builtin_amdgcn_wmma_f32_16x16x32_f16(
              false, qa[c].v, false, bb.v, (short)0, acc, false, false);
        }
        s[nt] = acc;
      }
      #pragma unroll
      for (int r = 0; r < 8; ++r) {
        const int qv = qlo + r + 8 * hl;
        s[0][r] = ((klo + l16)      <= qv) ? s[0][r] * scale : -2.38e38f;
        s[1][r] = ((klo + 16 + l16) <= qv) ? s[1][r] * scale : -2.38e38f;
      }
      float alpha[8], p0[8], p1[8];
      #pragma unroll
      for (int r = 0; r < 8; ++r) {
        float mt = rowmax16(fmaxf(s[0][r], s[1][r]));
        float mn = fmaxf(mrow[r], mt);
        alpha[r] = __expf(mrow[r] - mn);
        p0[r] = __expf(s[0][r] - mn);
        p1[r] = __expf(s[1][r] - mn);
        lrow[r] = lrow[r] * alpha[r] + rowsum16(p0[r] + p1[r]);
        mrow[r] = mn;
      }
      #pragma unroll
      for (int n = 0; n < 8; ++n)
        #pragma unroll
        for (int r = 0; r < 8; ++r) o[n][r] *= alpha[r];

      #pragma unroll
      for (int r = 0; r < 8; ++r) {
        const int pr = r + 8 * hl;
        sPw[pr * 32 + l16]      = (_Float16)p0[r];
        sPw[pr * 32 + 16 + l16] = (_Float16)p1[r];
      }
      asm volatile("s_wait_dscnt 0x0" ::: "memory");
      V16 pa;
      {
        const int base = hl * 8;
        pa.h[0] = *(const v8h*)&sPw[l16 * 32 + base];
        pa.h[1] = *(const v8h*)&sPw[l16 * 32 + base + 16];
      }
      #pragma unroll
      for (int n = 0; n < 8; ++n) {
        const int dcol = n * 16 + l16;
        const int kb2  = hl * 8;
        V16 vb;
        vb.h[0] = *(const v8h*)&sVt[dcol * VPAD + kb2];
        vb.h[1] = *(const v8h*)&sVt[dcol * VPAD + kb2 + 16];
        o[n] = __builtin_amdgcn_wmma_f32_16x16x32_f16(
            false, pa.v, false, vb.v, (short)0, o[n], false, false);
      }
    }
  }

  #pragma unroll
  for (int r = 0; r < 8; ++r) {
    const int srow = nc * C_ + qlo + r + 8 * hl;
    const float inv = 1.0f / lrow[r];
    float* op = out + (((size_t)b * S_ + srow) * HQ + hq) * D_;
    #pragma unroll
    for (int n = 0; n < 8; ++n) op[n * 16 + l16] = o[n][r] * inv;
  }
}

extern "C" void kernel_launch(void* const* d_in, const int* in_sizes, int n_in,
                              void* d_out, int out_size, void* d_ws, size_t ws_size,
                              hipStream_t stream) {
  (void)in_sizes; (void)n_in; (void)out_size;
  const float* q = (const float*)d_in[0];
  const float* k = (const float*)d_in[1];
  const float* v = (const float*)d_in[2];
  float* o = (float*)d_out;

  const size_t need = 2 * KV16_ELEMS * 2;  // K16 + VT16, 2 bytes each = 16 MB
  if (ws_size >= need) {
    _Float16* k16  = (_Float16*)d_ws;
    _Float16* vt16 = (_Float16*)((char*)d_ws + KV16_ELEMS * 2);
    hipLaunchKernelGGL(prep_kv_f16, dim3(B_ * NCH * HKV * 4), dim3(256), 0, stream,
                       k, v, k16, vt16);
    hipLaunchKernelGGL(chunk_attn_wmma_async, dim3(B_ * NCH * HQ * 4), dim3(256), 0, stream,
                       q, k16, vt16, o);
  } else {
    hipLaunchKernelGGL(chunk_attn_wmma_fused, dim3(B_ * NCH * HQ * 4), dim3(256), 0, stream,
                       q, k, v, o);
  }
}
